// PN2GeometryEncoderMSG_58463094833337
// MI455X (gfx1250) — compile-verified, matrix operand order checked by reading, not run on hardware
//
#include <hip/hip_runtime.h>
#include <hip/hip_bf16.h>

typedef __bf16 bf16;
typedef __attribute__((ext_vector_type(16))) __bf16 v16bf;
typedef __attribute__((ext_vector_type(8)))  __bf16 v8bf;
typedef __attribute__((ext_vector_type(8)))  float  v8f;

#define NBATCH 8
#define NPTS   8192
#define NS1    512
#define NS2    128

static inline int cdiv_h(int a, int b) { return (a + b - 1) / b; }
static inline int pad32_h(int k) { return (k + 31) & ~31; }

// ---------------------------------------------------------------------------
// Weight convert: f32 W[K,N] -> bf16 Wt[N,Kpad] (transposed, K zero-padded)
// ---------------------------------------------------------------------------
__global__ void convert_w_kernel(const float* __restrict__ W, bf16* __restrict__ Wt,
                                 int K, int N, int Kpad) {
  int i = blockIdx.x * blockDim.x + threadIdx.x;
  int tot = N * Kpad;
  if (i >= tot) return;
  int n = i / Kpad, kp = i % Kpad;
  Wt[(size_t)n * Kpad + kp] = (kp < K) ? (bf16)W[(size_t)kp * N + n] : (bf16)0.f;
}

// ---------------------------------------------------------------------------
// Farthest point sampling: one block per batch, dists live in registers.
// ---------------------------------------------------------------------------
template <int NP, int NSAMP, int TPB>
__global__ void fps_kernel(const float* __restrict__ pos, int* __restrict__ sidx,
                           float* __restrict__ spos) {
  __shared__ float smax[TPB];
  __shared__ int   sarg[TPB];
  __shared__ int   slast;
  const int t = threadIdx.x;
  constexpr int PPT = NP / TPB;
  float dist[PPT];
#pragma unroll
  for (int p = 0; p < PPT; ++p) dist[p] = 1e10f;
  if (t == 0) { slast = 0; sidx[0] = 0; }
  __syncthreads();
  for (int it = 1; it < NSAMP; ++it) {
    int last = slast;
    float lx = pos[last * 3 + 0], ly = pos[last * 3 + 1], lz = pos[last * 3 + 2];
    float best = -1.f; int bi = 0;
#pragma unroll
    for (int p = 0; p < PPT; ++p) {
      int i = p * TPB + t;
      float dx = pos[i * 3 + 0] - lx;
      float dy = pos[i * 3 + 1] - ly;
      float dz = pos[i * 3 + 2] - lz;
      float d  = dx * dx + dy * dy + dz * dz;
      float dd = fminf(dist[p], d);
      dist[p] = dd;
      if (dd > best) { best = dd; bi = i; }
    }
    smax[t] = best; sarg[t] = bi;
    __syncthreads();
    for (int s = TPB / 2; s > 0; s >>= 1) {
      if (t < s && smax[t + s] > smax[t]) { smax[t] = smax[t + s]; sarg[t] = sarg[t + s]; }
      __syncthreads();
    }
    if (t == 0) { slast = sarg[0]; sidx[it] = sarg[0]; }
    __syncthreads();
  }
  for (int i = t; i < NSAMP; i += TPB) {
    int src = sidx[i];
    spos[i * 3 + 0] = pos[src * 3 + 0];
    spos[i * 3 + 1] = pos[src * 3 + 1];
    spos[i * 3 + 2] = pos[src * 3 + 2];
  }
}

// ---------------------------------------------------------------------------
// Ball query: one wave32 per center; first-k-by-index via ballot scan.
// ---------------------------------------------------------------------------
__global__ void ballquery_kernel(const float* __restrict__ pos, int Np,
                                 const float* __restrict__ ctr, float r2, int k,
                                 int* __restrict__ nbr) {
  int s = blockIdx.x;
  int lane = threadIdx.x;
  float cx = ctr[s * 3 + 0], cy = ctr[s * 3 + 1], cz = ctr[s * 3 + 2];
  int cnt = 0;
  for (int base = 0; base < Np && cnt < k; base += 32) {
    int i = base + lane;
    bool in = false;
    if (i < Np) {
      float dx = pos[i * 3 + 0] - cx;
      float dy = pos[i * 3 + 1] - cy;
      float dz = pos[i * 3 + 2] - cz;
      in = (dx * dx + dy * dy + dz * dz) <= r2;
    }
    unsigned m = __builtin_amdgcn_ballot_w32(in);
    int rank = __popc(m & ((1u << lane) - 1u));
    if (in && (cnt + rank) < k) nbr[s * k + cnt + rank] = i;
    cnt += __popc(m);
  }
  if (cnt > k) cnt = k;
  for (int j = cnt + lane; j < k; j += 32) nbr[s * k + j] = -1;
}

// ---------------------------------------------------------------------------
// Gather kernels: build bf16 feature matrices [rows, Kpad], zero padded.
// ---------------------------------------------------------------------------
__global__ void gather_sa1_kernel(const float* __restrict__ pos,
                                  const float* __restrict__ ctr,
                                  const int* __restrict__ nbr, int k, int M,
                                  bf16* __restrict__ A) {
  int row = blockIdx.x * blockDim.x + threadIdx.x;
  if (row >= M) return;
  bf16* a = A + (size_t)row * 32;
  int idx = nbr[row];
  if (idx >= 0) {
    int s = row / k;
    float px = pos[idx * 3 + 0], py = pos[idx * 3 + 1], pz = pos[idx * 3 + 2];
    a[0] = (bf16)px; a[1] = (bf16)py; a[2] = (bf16)pz;
    a[3] = (bf16)(px - ctr[s * 3 + 0]);
    a[4] = (bf16)(py - ctr[s * 3 + 1]);
    a[5] = (bf16)(pz - ctr[s * 3 + 2]);
    for (int c = 6; c < 32; ++c) a[c] = (bf16)0.f;
  } else {
    for (int c = 0; c < 32; ++c) a[c] = (bf16)0.f;
  }
}

__global__ void gather_sa2_kernel(const bf16* __restrict__ X,
                                  const float* __restrict__ pos,
                                  const float* __restrict__ ctr,
                                  const int* __restrict__ nbr, int k, int M,
                                  bf16* __restrict__ A) {
  int row = blockIdx.x * blockDim.x + threadIdx.x;
  if (row >= M) return;
  bf16* a = A + (size_t)row * 352;
  int idx = nbr[row];
  if (idx < 0) {
    for (int c = 0; c < 352; ++c) a[c] = (bf16)0.f;
    return;
  }
  int s = row / k;
  const bf16* xr = X + (size_t)idx * 320;
  for (int c = 0; c < 320; ++c) a[c] = xr[c];
  a[320] = (bf16)(pos[idx * 3 + 0] - ctr[s * 3 + 0]);
  a[321] = (bf16)(pos[idx * 3 + 1] - ctr[s * 3 + 1]);
  a[322] = (bf16)(pos[idx * 3 + 2] - ctr[s * 3 + 2]);
  for (int c = 323; c < 352; ++c) a[c] = (bf16)0.f;
}

// ---------------------------------------------------------------------------
// WMMA GEMM + (BN+ReLU | bias) epilogue, register-blocked in N.
// A: bf16 [Mpad, K] row-major, K multiple of 32, Mpad multiple of 64.
// Wt: bf16 [N, K] row-major (transposed weights).
// Block = 128 threads (4 waves); wave computes 16 x (16*NT). All NT B
// fragments are preloaded into distinct registers before the WMMA chain so
// there is no WAR register aliasing between the next tile's loads and the
// current WMMA -> partial s_wait_loadcnt overlap instead of full drains.
// ---------------------------------------------------------------------------
template <int NT>
__global__ void wmma_gemm_kernel(const bf16* __restrict__ A, int K,
                                 const bf16* __restrict__ Wt,
                                 const float* __restrict__ gamma,
                                 const float* __restrict__ beta,
                                 const float* __restrict__ bias, int relu,
                                 bf16* __restrict__ outb,
                                 float* __restrict__ outf, int Ncols, int Mreal) {
  const int lane = threadIdx.x & 31;
  const int wave = threadIdx.x >> 5;
  const int rowbase = blockIdx.x * 64 + wave * 16;
  const int coltile0 = blockIdx.y * (16 * NT);
  const int hl = lane & 15;
  const int koff = (lane < 16) ? 0 : 8;

  const bf16* ap = A + (size_t)(rowbase + hl) * K + koff;
  const bf16* bbase = Wt + (size_t)(coltile0 + hl) * K + koff;
  const size_t bstride = (size_t)16 * K;  // elements between adjacent col tiles

  v8f acc[NT];
#pragma unroll
  for (int t = 0; t < NT; ++t) acc[t] = {};

  const int ksteps = K >> 5;
  for (int ks = 0; ks < ksteps; ++ks) {
    // prefetch next K-panel of the activation stream (global_prefetch_b8)
    __builtin_prefetch(ap + 32, 0, 1);
    v8bf alo = *(const v8bf*)(ap);
    v8bf ahi = *(const v8bf*)(ap + 16);
    v16bf af = __builtin_shufflevector(alo, ahi, 0, 1, 2, 3, 4, 5, 6, 7,
                                       8, 9, 10, 11, 12, 13, 14, 15);
    const bf16* bk = bbase + (size_t)ks * 32;

    // Preload all B fragments into distinct registers (no WAR aliasing).
    v16bf bfrag[NT];
#pragma unroll
    for (int t = 0; t < NT; ++t) {
      v8bf blo = *(const v8bf*)(bk + (size_t)t * bstride);
      v8bf bhi = *(const v8bf*)(bk + (size_t)t * bstride + 16);
      bfrag[t] = __builtin_shufflevector(blo, bhi, 0, 1, 2, 3, 4, 5, 6, 7,
                                         8, 9, 10, 11, 12, 13, 14, 15);
    }
#pragma unroll
    for (int t = 0; t < NT; ++t) {
      acc[t] = __builtin_amdgcn_wmma_f32_16x16x32_bf16(false, af, false, bfrag[t],
                                                       (short)0, acc[t], false, false);
    }
    ap += 32;
  }

#pragma unroll
  for (int t = 0; t < NT; ++t) {
    const int n = coltile0 + t * 16 + hl;
    float g = gamma ? gamma[n] : 1.f;
    float bb = beta ? beta[n] : (bias ? bias[n] : 0.f);
#pragma unroll
    for (int r = 0; r < 8; ++r) {
      int m = rowbase + r + ((lane >= 16) ? 8 : 0);
      float v = acc[t][r] * g + bb;
      if (relu) v = fmaxf(v, 0.f);
      if (outb) outb[(size_t)m * Ncols + n] = (bf16)v;
      if (outf && m < Mreal) outf[(size_t)m * Ncols + n] = v;
    }
  }
}

// ---------------------------------------------------------------------------
// Masked max-aggregation over k neighbors per center.
// ---------------------------------------------------------------------------
__global__ void max_agg_kernel(const bf16* __restrict__ H, const int* __restrict__ nbr,
                               int k, int Cm, bf16* __restrict__ out, int ldo,
                               int colofs) {
  int s = blockIdx.x;
  int c = threadIdx.x;
  if (c >= Cm) return;
  float m = -1e30f;
  bool any = false;
  const bf16* hp = H + (size_t)s * k * Cm + c;
  const int* np = nbr + s * k;
  for (int j = 0; j < k; ++j) {
    if (np[j] >= 0) {
      any = true;
      m = fmaxf(m, (float)hp[(size_t)j * Cm]);
    }
  }
  out[(size_t)s * ldo + colofs + c] = (bf16)(any ? m : 0.f);
}

// ---------------------------------------------------------------------------
// Global max pool over rows -> row 0 of A (rest pre-zeroed).
// ---------------------------------------------------------------------------
__global__ void maxpool_kernel(const bf16* __restrict__ X, int rows, int C,
                               bf16* __restrict__ out) {
  int c = blockIdx.x * blockDim.x + threadIdx.x;
  if (c >= C) return;
  float m = -1e30f;
  for (int r = 0; r < rows; ++r) m = fmaxf(m, (float)X[(size_t)r * C + c]);
  out[c] = (bf16)m;
}

// ---------------------------------------------------------------------------
// 3-NN inverse-squared-distance interpolation.
// ---------------------------------------------------------------------------
__global__ void knn3_kernel(const float* __restrict__ pd, int Nd,
                            const float* __restrict__ ps, int Ns,
                            const bf16* __restrict__ X, int C,
                            bf16* __restrict__ A, int lda, int colofs) {
  int i = blockIdx.x * blockDim.x + threadIdx.x;
  if (i >= Nd) return;
  float x = pd[i * 3 + 0], y = pd[i * 3 + 1], z = pd[i * 3 + 2];
  float d0 = 1e30f, d1 = 1e30f, d2 = 1e30f;
  int i0 = 0, i1 = 0, i2 = 0;
  for (int j = 0; j < Ns; ++j) {
    float dx = ps[j * 3 + 0] - x;
    float dy = ps[j * 3 + 1] - y;
    float dz = ps[j * 3 + 2] - z;
    float d = dx * dx + dy * dy + dz * dz;
    if (d < d0)      { d2 = d1; i2 = i1; d1 = d0; i1 = i0; d0 = d; i0 = j; }
    else if (d < d1) { d2 = d1; i2 = i1; d1 = d;  i1 = j; }
    else if (d < d2) { d2 = d;  i2 = j; }
  }
  float w0 = 1.f / fmaxf(d0, 1e-16f);
  float w1 = 1.f / fmaxf(d1, 1e-16f);
  float w2 = 1.f / fmaxf(d2, 1e-16f);
  float ws = w0 + w1 + w2;
  const bf16* r0 = X + (size_t)i0 * C;
  const bf16* r1 = X + (size_t)i1 * C;
  const bf16* r2 = X + (size_t)i2 * C;
  bf16* out = A + (size_t)i * lda + colofs;
  for (int c = 0; c < C; ++c)
    out[c] = (bf16)((w0 * (float)r0[c] + w1 * (float)r1[c] + w2 * (float)r2[c]) / ws);
}

// ---------------------------------------------------------------------------
// Utility: zero / concat copies.
// ---------------------------------------------------------------------------
__global__ void zero_bf16_kernel(bf16* __restrict__ p, int n) {
  int i = blockIdx.x * blockDim.x + threadIdx.x;
  if (i < n) p[i] = (bf16)0.f;
}

__global__ void copy_bf16_cols_kernel(const bf16* __restrict__ src, int C,
                                      bf16* __restrict__ A, int lda, int colofs,
                                      int total) {
  int i = blockIdx.x * blockDim.x + threadIdx.x;
  if (i >= total) return;
  int row = i / C, c = i % C;
  A[(size_t)row * lda + colofs + c] = src[i];
}

__global__ void copy_pos_cols_kernel(const float* __restrict__ pos,
                                     bf16* __restrict__ A, int lda, int colofs,
                                     int total) {
  int i = blockIdx.x * blockDim.x + threadIdx.x;
  if (i >= total) return;
  int row = i / 3, c = i % 3;
  A[(size_t)row * lda + colofs + c] = (bf16)pos[i];
}

// ---------------------------------------------------------------------------
// Host orchestration
// ---------------------------------------------------------------------------
extern "C" void kernel_launch(void* const* d_in, const int* in_sizes, int n_in,
                              void* d_out, int out_size, void* d_ws, size_t ws_size,
                              hipStream_t stream) {
  (void)in_sizes; (void)n_in; (void)out_size; (void)ws_size;
  auto P = [&](int i) { return (const float*)d_in[i]; };

  // Channel tables
  static const int SA1CH[3][4] = {{6, 32, 32, 64}, {6, 64, 64, 128}, {6, 64, 96, 128}};
  static const int SA2CH[3][4] = {{323, 64, 64, 128}, {323, 128, 128, 256}, {323, 128, 128, 256}};
  static const int   SA1K[3]  = {16, 32, 128};
  static const float SA1R2[3] = {0.01f, 0.04f, 0.16f};
  static const int   SA2K[3]  = {32, 64, 128};
  static const float SA2R2[3] = {0.04f, 0.16f, 0.64f};
  static const int C1OFS[3] = {0, 64, 192};   // C1 = 320
  static const int C2OFS[3] = {0, 128, 384};  // C2 = 640

  // Weight table: d_in index, K, N (order matches usage below)
  struct WInfo { int pidx, K, N; };
  WInfo WT[24]; int nw = 0;
  for (int br = 0; br < 3; ++br)
    for (int l = 0; l < 3; ++l) WT[nw++] = {1 + 9 * br + 3 * l, SA1CH[br][l], SA1CH[br][l + 1]};
  for (int br = 0; br < 3; ++br)
    for (int l = 0; l < 3; ++l) WT[nw++] = {28 + 9 * br + 3 * l, SA2CH[br][l], SA2CH[br][l + 1]};
  WT[nw++] = {55, 640, 512}; WT[nw++] = {58, 512, 256};   // glob
  WT[nw++] = {60, 960, 256}; WT[nw++] = {63, 256, 256};   // fp1
  WT[nw++] = {66, 259, 256}; WT[nw++] = {69, 256, 256};   // fp0

  size_t woff[24], wtot = 0;
  for (int i = 0; i < 24; ++i) { woff[i] = wtot; wtot += (size_t)WT[i].N * pad32_h(WT[i].K); }

  // Workspace carve-out (256B aligned)
  char* wsp = (char*)d_ws;
  size_t off = 0;
  auto take = [&](size_t bytes) -> void* {
    void* p = wsp + off;
    off = (off + bytes + 255) & ~(size_t)255;
    return p;
  };
  bf16*  wts  = (bf16*)take(wtot * sizeof(bf16));
  int*   idx1 = (int*)take(NS1 * sizeof(int));
  float* pos1 = (float*)take(NS1 * 3 * sizeof(float));
  int*   idx2 = (int*)take(NS2 * sizeof(int));
  float* pos2 = (float*)take(NS2 * 3 * sizeof(float));
  int*   nbr  = (int*)take((size_t)NS1 * 128 * sizeof(int));
  bf16*  x1   = (bf16*)take((size_t)NS1 * 320 * sizeof(bf16));
  bf16*  x2   = (bf16*)take((size_t)NS2 * 640 * sizeof(bf16));
  bf16*  x1fp = (bf16*)take((size_t)NS1 * 256 * sizeof(bf16));
  bf16*  Abuf = (bf16*)take((size_t)16384 * 352 * sizeof(bf16));     // largest A
  bf16*  H0   = (bf16*)take((size_t)65536 * 128 * sizeof(bf16));     // largest hidden
  bf16*  H1   = (bf16*)take((size_t)65536 * 128 * sizeof(bf16));

  float* Fout = (float*)d_out;                               // [8, 8192, 256]
  float* gout = Fout + (size_t)NBATCH * NPTS * 256;          // [8, 256]

  // Convert all weights to transposed, K-padded bf16 (deterministic each call)
  for (int i = 0; i < 24; ++i) {
    int Kp = pad32_h(WT[i].K);
    int tot = WT[i].N * Kp;
    convert_w_kernel<<<cdiv_h(tot, 256), 256, 0, stream>>>(P(WT[i].pidx), wts + woff[i],
                                                           WT[i].K, WT[i].N, Kp);
  }

  auto gemm = [&](const bf16* A, int Kpad, int widx, const float* gamma,
                  const float* beta, const float* bias, int relu, bf16* outb,
                  float* outf, int N, int Mpad, int Mreal) {
    if (N % 64 == 0) {
      dim3 grid(Mpad / 64, N / 64);
      wmma_gemm_kernel<4><<<grid, 128, 0, stream>>>(A, Kpad, wts + woff[widx], gamma,
                                                    beta, bias, relu, outb, outf, N, Mreal);
    } else {  // N % 32 == 0 (e.g. 32, 96)
      dim3 grid(Mpad / 64, N / 32);
      wmma_gemm_kernel<2><<<grid, 128, 0, stream>>>(A, Kpad, wts + woff[widx], gamma,
                                                    beta, bias, relu, outb, outf, N, Mreal);
    }
  };

  for (int b = 0; b < NBATCH; ++b) {
    const float* pos = P(0) + (size_t)b * NPTS * 3;

    // ----- SA1 -----
    fps_kernel<NPTS, NS1, 1024><<<1, 1024, 0, stream>>>(pos, idx1, pos1);
    for (int br = 0; br < 3; ++br) {
      int k = SA1K[br], M = NS1 * k;  // multiple of 64
      ballquery_kernel<<<NS1, 32, 0, stream>>>(pos, NPTS, pos1, SA1R2[br], k, nbr);
      gather_sa1_kernel<<<cdiv_h(M, 256), 256, 0, stream>>>(pos, pos1, nbr, k, M, Abuf);
      const bf16* cur = Abuf;
      int curK = 32;
      for (int l = 0; l < 3; ++l) {
        int pb = 1 + 9 * br + 3 * l;
        bf16* ob = (l & 1) ? H1 : H0;
        gemm(cur, curK, 3 * br + l, P(pb + 1), P(pb + 2), nullptr, 1, ob, nullptr,
             SA1CH[br][l + 1], M, M);
        cur = ob;
        curK = SA1CH[br][l + 1];
      }
      max_agg_kernel<<<NS1, SA1CH[br][3], 0, stream>>>(cur, nbr, k, SA1CH[br][3],
                                                       x1, 320, C1OFS[br]);
    }

    // ----- SA2 -----
    fps_kernel<NS1, NS2, 512><<<1, 512, 0, stream>>>(pos1, idx2, pos2);
    for (int br = 0; br < 3; ++br) {
      int k = SA2K[br], M = NS2 * k;  // multiple of 64
      ballquery_kernel<<<NS2, 32, 0, stream>>>(pos1, NS1, pos2, SA2R2[br], k, nbr);
      gather_sa2_kernel<<<cdiv_h(M, 256), 256, 0, stream>>>(x1, pos1, pos2, nbr, k, M, Abuf);
      const bf16* cur = Abuf;
      int curK = 352;
      for (int l = 0; l < 3; ++l) {
        int pb = 28 + 9 * br + 3 * l;
        bf16* ob = (l & 1) ? H1 : H0;
        gemm(cur, curK, 9 + 3 * br + l, P(pb + 1), P(pb + 2), nullptr, 1, ob, nullptr,
             SA2CH[br][l + 1], M, M);
        cur = ob;
        curK = SA2CH[br][l + 1];
      }
      max_agg_kernel<<<NS2, SA2CH[br][3], 0, stream>>>(cur, nbr, k, SA2CH[br][3],
                                                       x2, 640, C2OFS[br]);
    }

    // ----- Global MLP: maxpool(x2) -> 512 -> 256 -----
    zero_bf16_kernel<<<cdiv_h(64 * 640, 256), 256, 0, stream>>>(Abuf, 64 * 640);
    maxpool_kernel<<<cdiv_h(640, 128), 128, 0, stream>>>(x2, NS2, 640, Abuf);
    gemm(Abuf, 640, 18, P(56), P(57), nullptr, 1, H0, nullptr, 512, 64, 64);
    gemm(H0, 512, 19, nullptr, nullptr, P(59), 0, nullptr, gout + (size_t)b * 256, 256, 64, 1);

    // ----- FP1: knn(x2 -> pos1) ++ x1 -> 256 -> 256 -----
    zero_bf16_kernel<<<cdiv_h(NS1 * 960, 256), 256, 0, stream>>>(Abuf, NS1 * 960);
    knn3_kernel<<<cdiv_h(NS1, 128), 128, 0, stream>>>(pos1, NS1, pos2, NS2, x2, 640,
                                                      Abuf, 960, 0);
    copy_bf16_cols_kernel<<<cdiv_h(NS1 * 320, 256), 256, 0, stream>>>(x1, 320, Abuf, 960,
                                                                      640, NS1 * 320);
    gemm(Abuf, 960, 20, P(61), P(62), nullptr, 1, H0, nullptr, 256, NS1, NS1);
    gemm(H0, 256, 21, P(64), P(65), nullptr, 1, x1fp, nullptr, 256, NS1, NS1);

    // ----- FP0: knn(x1fp -> pos) ++ pos -> 256 -> 256 -> F -----
    zero_bf16_kernel<<<cdiv_h(NPTS * 288, 256), 256, 0, stream>>>(Abuf, NPTS * 288);
    knn3_kernel<<<cdiv_h(NPTS, 128), 128, 0, stream>>>(pos, NPTS, pos1, NS1, x1fp, 256,
                                                       Abuf, 288, 0);
    copy_pos_cols_kernel<<<cdiv_h(NPTS * 3, 256), 256, 0, stream>>>(pos, Abuf, 288, 256,
                                                                    NPTS * 3);
    gemm(Abuf, 288, 22, P(67), P(68), nullptr, 1, H0, nullptr, 256, NPTS, NPTS);
    gemm(H0, 256, 23, nullptr, nullptr, P(70), 0, nullptr,
         Fout + (size_t)b * NPTS * 256, 256, NPTS, NPTS);
  }
}
